// SwinTransformerLayer_64828236366109
// MI455X (gfx1250) — compile-verified, matrix operand order checked by reading, not run on hardware
//
#include <hip/hip_runtime.h>
#include <hip/hip_bf16.h>

// ---------------------------------------------------------------------------
// Swin-3D (Pangu-style) transformer layer x2 for MI455X (gfx1250, wave32).
// All matmuls run on V_WMMA_F32_16X16X32_F16 with fp32 accumulation.
// ---------------------------------------------------------------------------

typedef __attribute__((ext_vector_type(16))) _Float16 v16h;
typedef __attribute__((ext_vector_type(8)))  float    v8f;

#define Zd 8
#define Hd 96
#define Wd 192
#define WZ 2
#define WH 6
#define WW 12
#define DIMc 192
#define HEADS 6
#define HD 32
#define NZ 4
#define NH 16
#define NW 16
#define NTOK 144            // tokens per window
#define NWIN 1024           // windows
#define MW 147456           // total tokens (= NWIN*NTOK)
#define WT 64               // window types
#define VPAD 160            // K=144 padded to 160 for WMMA K-steps of 32
#define SCALEQ 0.17677669529663687f

static __device__ __forceinline__ int wrow_to_token(int wrow, int shifted) {
    int wid = wrow / NTOK, t = wrow - wid * NTOK;
    int nz = wid >> 8, nh = (wid >> 4) & 15, nw = wid & 15;
    int z = t / 72, rem = t - z * 72, h = rem / 12, w = rem - h * 12;
    int pz = nz * WZ + z, ph = nh * WH + h, pw = nw * WW + w;
    if (shifted) {
        pz = (pz + 1) & 7;
        ph += 3; if (ph >= Hd) ph -= Hd;
        pw += 6; if (pw >= Wd) pw -= Wd;
    }
    return (pz * Hd + ph) * Wd + pw;
}

static __device__ __forceinline__ int region_of(int za, int ha, int wa) {
    int rz = (za < 6) ? 0 : ((za < 7) ? 1 : 2);
    int rh = (ha < 90) ? 0 : ((ha < 93) ? 1 : 2);
    int rw = (wa < 180) ? 0 : ((wa < 186) ? 1 : 2);
    return rz * 9 + rh * 3 + rw;
}

// --------------------------- weight transpose ------------------------------
// out[n*K + k] = (f16) in[k*N + n]   (stores W^T so B-fragments are contiguous)
__global__ void __launch_bounds__(256) transpose_to_f16(
    const float* __restrict__ in, _Float16* __restrict__ out, int K, int N) {
    int idx = blockIdx.x * 256 + threadIdx.x;
    if (idx >= K * N) return;
    int n = idx / K, k = idx - n * K;
    out[idx] = (_Float16)in[(size_t)k * N + n];
}

// ------------------------------- LayerNorm ---------------------------------
// one wave32 per row of 192; mode=1: output row is a windowed row, source is
// the rolled token (fused roll + partition); mode=0: identity row mapping.
__global__ void __launch_bounds__(256) ln_kernel(
    const float* __restrict__ x, const float* __restrict__ sw,
    const float* __restrict__ sb, _Float16* __restrict__ out,
    int mode, int shifted) {
    int row  = blockIdx.x * 8 + (threadIdx.x >> 5);
    int lane = threadIdx.x & 31;
    int src  = mode ? wrow_to_token(row, shifted) : row;
    const float* xp = x + (size_t)src * DIMc;
    float v[6];
    float sum = 0.f;
#pragma unroll
    for (int k = 0; k < 6; ++k) { v[k] = xp[lane + k * 32]; sum += v[k]; }
    sum += __shfl_xor(sum, 16, 32); sum += __shfl_xor(sum, 8, 32);
    sum += __shfl_xor(sum, 4, 32);  sum += __shfl_xor(sum, 2, 32);
    sum += __shfl_xor(sum, 1, 32);
    float mean = sum * (1.0f / 192.0f);
    float var = 0.f;
#pragma unroll
    for (int k = 0; k < 6; ++k) { float d = v[k] - mean; var += d * d; }
    var += __shfl_xor(var, 16, 32); var += __shfl_xor(var, 8, 32);
    var += __shfl_xor(var, 4, 32);  var += __shfl_xor(var, 2, 32);
    var += __shfl_xor(var, 1, 32);
    float rs = rsqrtf(var * (1.0f / 192.0f) + 1e-5f);
    _Float16* op = out + (size_t)row * DIMc;
#pragma unroll
    for (int k = 0; k < 6; ++k) {
        int c = lane + k * 32;
        op[c] = (_Float16)((v[k] - mean) * rs * sw[c] + sb[c]);
    }
}

// ------------------------------ WMMA GEMM core -----------------------------
// Block: 256 threads = 8 waves, tile 128(M) x 64(N); wave tile 32x32.
// A: MxK f16 row-major (stride K). Bt: NxK f16 row-major (= W^T).
static __device__ __forceinline__ void wmma_gemm_32x32(
    const _Float16* __restrict__ A, const _Float16* __restrict__ Bt,
    int K, size_t row0, size_t col0, v8f (&acc)[2][2]) {
    int lane = threadIdx.x & 31;
    int mrow = lane & 15;
    int kof  = (lane >> 4) * 16;
    const _Float16* a0p = A  + (row0 + mrow) * (size_t)K + kof;
    const _Float16* a1p = a0p + (size_t)16 * K;
    const _Float16* b0p = Bt + (col0 + mrow) * (size_t)K + kof;
    const _Float16* b1p = b0p + (size_t)16 * K;
    for (int k0 = 0; k0 < K; k0 += 32) {
        v16h a0 = *(const v16h*)(a0p + k0);
        v16h a1 = *(const v16h*)(a1p + k0);
        v16h b0 = *(const v16h*)(b0p + k0);
        v16h b1 = *(const v16h*)(b1p + k0);
        acc[0][0] = __builtin_amdgcn_wmma_f32_16x16x32_f16(false, a0, false, b0, (short)0, acc[0][0], false, false);
        acc[0][1] = __builtin_amdgcn_wmma_f32_16x16x32_f16(false, a0, false, b1, (short)0, acc[0][1], false, false);
        acc[1][0] = __builtin_amdgcn_wmma_f32_16x16x32_f16(false, a1, false, b0, (short)0, acc[1][0], false, false);
        acc[1][1] = __builtin_amdgcn_wmma_f32_16x16x32_f16(false, a1, false, b1, (short)0, acc[1][1], false, false);
    }
}

#define GEMM_PROLOG(BN)                                            \
    int lane = threadIdx.x & 31;                                   \
    int wave = threadIdx.x >> 5;                                   \
    size_t row0 = (size_t)blockIdx.x * 128 + (wave & 3) * 32;      \
    size_t col0 = (size_t)blockIdx.y * 64 + (wave >> 2) * 32;      \
    v8f acc[2][2] = {{{}, {}}, {{}, {}}};

// QKV: N=576, K=192. Writes Q (scaled), K rows, and V transposed+padded.
__global__ void __launch_bounds__(256) gemm_qkv(
    const _Float16* __restrict__ A, const _Float16* __restrict__ Bt,
    const float* __restrict__ bias, _Float16* __restrict__ Qo,
    _Float16* __restrict__ Ko, _Float16* __restrict__ Vt) {
    GEMM_PROLOG(576)
    wmma_gemm_32x32(A, Bt, DIMc, row0, col0, acc);
#pragma unroll
    for (int mt = 0; mt < 2; ++mt)
#pragma unroll
    for (int nt = 0; nt < 2; ++nt)
#pragma unroll
    for (int r = 0; r < 8; ++r) {
        size_t m = row0 + mt * 16 + (lane >> 4) * 8 + r;
        int n = (int)col0 + nt * 16 + (lane & 15);
        float v = acc[mt][nt][r] + bias[n];
        int qi = n / DIMc, rem = n - qi * DIMc;   // rem = head*32 + d
        if (qi == 0) {
            Qo[m * DIMc + rem] = (_Float16)(v * SCALEQ);
        } else if (qi == 1) {
            Ko[m * DIMc + rem] = (_Float16)v;
        } else {
            int wid = (int)(m / NTOK), tok = (int)(m - (size_t)wid * NTOK);
            Vt[((size_t)wid * DIMc + rem) * VPAD + tok] = (_Float16)v;
        }
    }
}

// Proj: N=192, K=192. Un-partitions (+un-roll) and adds residual -> fp32.
__global__ void __launch_bounds__(256) gemm_proj(
    const _Float16* __restrict__ A, const _Float16* __restrict__ Bt,
    const float* __restrict__ bias, const float* __restrict__ resid,
    float* __restrict__ out, int shifted) {
    GEMM_PROLOG(192)
    wmma_gemm_32x32(A, Bt, DIMc, row0, col0, acc);
#pragma unroll
    for (int mt = 0; mt < 2; ++mt)
#pragma unroll
    for (int nt = 0; nt < 2; ++nt)
#pragma unroll
    for (int r = 0; r < 8; ++r) {
        size_t m = row0 + mt * 16 + (lane >> 4) * 8 + r;
        int n = (int)col0 + nt * 16 + (lane & 15);
        int tok = wrow_to_token((int)m, shifted);
        float v = acc[mt][nt][r] + bias[n] + resid[(size_t)tok * DIMc + n];
        out[(size_t)tok * DIMc + n] = v;
    }
}

// MLP1: N=768, K=192, exact GELU, f16 out.
__global__ void __launch_bounds__(256) gemm_gelu(
    const _Float16* __restrict__ A, const _Float16* __restrict__ Bt,
    const float* __restrict__ bias, _Float16* __restrict__ out) {
    GEMM_PROLOG(768)
    wmma_gemm_32x32(A, Bt, DIMc, row0, col0, acc);
#pragma unroll
    for (int mt = 0; mt < 2; ++mt)
#pragma unroll
    for (int nt = 0; nt < 2; ++nt)
#pragma unroll
    for (int r = 0; r < 8; ++r) {
        size_t m = row0 + mt * 16 + (lane >> 4) * 8 + r;
        int n = (int)col0 + nt * 16 + (lane & 15);
        float v = acc[mt][nt][r] + bias[n];
        float g = 0.5f * v * (1.0f + erff(v * 0.70710678118654752f));
        out[m * 768 + n] = (_Float16)g;
    }
}

// MLP2: N=192, K=768, bias + residual -> fp32 out (token order).
__global__ void __launch_bounds__(256) gemm_resid(
    const _Float16* __restrict__ A, const _Float16* __restrict__ Bt,
    const float* __restrict__ bias, const float* __restrict__ resid,
    float* __restrict__ out) {
    GEMM_PROLOG(192)
    wmma_gemm_32x32(A, Bt, 768, row0, col0, acc);
#pragma unroll
    for (int mt = 0; mt < 2; ++mt)
#pragma unroll
    for (int nt = 0; nt < 2; ++nt)
#pragma unroll
    for (int r = 0; r < 8; ++r) {
        size_t m = row0 + mt * 16 + (lane >> 4) * 8 + r;
        int n = (int)col0 + nt * 16 + (lane & 15);
        out[m * DIMc + n] = acc[mt][nt][r] + bias[n] + resid[m * DIMc + n];
    }
}

// ------------------------------- Attention ---------------------------------
// One wave per (window, head, 16-row block).  S = qk^T via 9 WMMAs (K=32),
// earth-position bias gather + shift mask computed on the fly, fp32 softmax
// with 16-lane shuffles, P staged in LDS (stride 160, zero padded), then
// O = P V via 10 WMMAs (K padded to 160).
__global__ void __launch_bounds__(32) attn_kernel(
    const _Float16* __restrict__ Q, const _Float16* __restrict__ Kb,
    const _Float16* __restrict__ Vt, _Float16* __restrict__ AO,
    const float* __restrict__ bias_tab, int masked) {
    __shared__ __align__(32) _Float16 P[16 * VPAD];
    int lane = threadIdx.x & 31;
    int mrow = lane & 15;
    int kh   = lane >> 4;                 // 0/1: K-half / row-half select
    int rb   = blockIdx.x;                // 0..8  row block
    int h    = blockIdx.y;                // 0..5  head
    int wid  = blockIdx.z;                // 0..1023 window
    int wt   = wid >> 4;
    int nz = wid >> 8, nh = (wid >> 4) & 15, nw = wid & 15;

    size_t qrow = (size_t)wid * NTOK + rb * 16 + mrow;
    v16h afrag = *(const v16h*)(Q + qrow * DIMc + h * HD + kh * 16);

    v8f s[9];
#pragma unroll
    for (int nt = 0; nt < 9; ++nt) {
        size_t krow = (size_t)wid * NTOK + nt * 16 + mrow;
        v16h bfrag = *(const v16h*)(Kb + krow * DIMc + h * HD + kh * 16);
        v8f z = {};
        s[nt] = __builtin_amdgcn_wmma_f32_16x16x32_f16(false, afrag, false, bfrag, (short)0, z, false, false);
    }

    // bias + mask
#pragma unroll
    for (int nt = 0; nt < 9; ++nt) {
        int j  = nt * 16 + mrow;
        int zj = j / 72, rj = j - zj * 72, hj = rj / 12, wj = rj - hj * 12;
        int regj = 0;
        if (masked) regj = region_of(nz * WZ + zj, nh * WH + hj, nw * WW + wj);
#pragma unroll
        for (int r = 0; r < 8; ++r) {
            int i  = rb * 16 + kh * 8 + r;
            int zi = i / 72, ri = i - zi * 72, hi = ri / 12, wi = ri - hi * 12;
            int pos = 828 * (zi + 2 * zj) + 23 * (hi + 6 * hj) + (wi - wj + 11);
            float bv = bias_tab[(size_t)pos * (WT * HEADS) + wt * HEADS + h];
            if (masked) {
                int regi = region_of(nz * WZ + zi, nh * WH + hi, nw * WW + wi);
                if (regi != regj) bv -= 100.0f;
            }
            s[nt][r] += bv;
        }
    }

    // softmax (rows r / 8+r live in lane halves; reduce across 16 lanes)
#pragma unroll
    for (int r = 0; r < 8; ++r) {
        float mx = -1e30f;
#pragma unroll
        for (int nt = 0; nt < 9; ++nt) mx = fmaxf(mx, s[nt][r]);
        mx = fmaxf(mx, __shfl_xor(mx, 8, 32));
        mx = fmaxf(mx, __shfl_xor(mx, 4, 32));
        mx = fmaxf(mx, __shfl_xor(mx, 2, 32));
        mx = fmaxf(mx, __shfl_xor(mx, 1, 32));
        float sum = 0.f;
#pragma unroll
        for (int nt = 0; nt < 9; ++nt) {
            float e = __expf(s[nt][r] - mx);
            s[nt][r] = e;
            sum += e;
        }
        sum += __shfl_xor(sum, 8, 32);
        sum += __shfl_xor(sum, 4, 32);
        sum += __shfl_xor(sum, 2, 32);
        sum += __shfl_xor(sum, 1, 32);
        float inv = 1.0f / sum;
#pragma unroll
        for (int nt = 0; nt < 9; ++nt)
            P[(kh * 8 + r) * VPAD + nt * 16 + mrow] = (_Float16)(s[nt][r] * inv);
    }
    // zero the K-pad columns 144..159
#pragma unroll
    for (int q = 0; q < 8; ++q)
        P[mrow * VPAD + NTOK + kh * 8 + q] = (_Float16)0.0f;
    asm volatile("s_wait_dscnt 0" ::: "memory");

    // O = P @ V  (M=16, N=32, Kpad=160)
    v8f o0 = {}, o1 = {};
    const _Float16* vtb = Vt + ((size_t)wid * HEADS + h) * HD * VPAD;
#pragma unroll
    for (int kt = 0; kt < 5; ++kt) {
        v16h pa = *(const v16h*)(&P[mrow * VPAD + kt * 32 + kh * 16]);
        v16h b0 = *(const v16h*)(vtb + (size_t)mrow * VPAD + kt * 32 + kh * 16);
        v16h b1 = *(const v16h*)(vtb + (size_t)(16 + mrow) * VPAD + kt * 32 + kh * 16);
        o0 = __builtin_amdgcn_wmma_f32_16x16x32_f16(false, pa, false, b0, (short)0, o0, false, false);
        o1 = __builtin_amdgcn_wmma_f32_16x16x32_f16(false, pa, false, b1, (short)0, o1, false, false);
    }
#pragma unroll
    for (int r = 0; r < 8; ++r) {
        size_t orow = (size_t)wid * NTOK + rb * 16 + kh * 8 + r;
        AO[orow * DIMc + h * HD + mrow]      = (_Float16)o0[r];
        AO[orow * DIMc + h * HD + 16 + mrow] = (_Float16)o1[r];
    }
}

// ------------------------------- launcher ----------------------------------
extern "C" void kernel_launch(void* const* d_in, const int* in_sizes, int n_in,
                              void* d_out, int out_size, void* d_ws, size_t ws_size,
                              hipStream_t stream) {
    (void)in_sizes; (void)n_in; (void)out_size; (void)ws_size;
    const float* x    = (const float*)d_in[0];
    const float* n1w  = (const float*)d_in[1];
    const float* n1b  = (const float*)d_in[2];
    const float* qkvw = (const float*)d_in[3];
    const float* qkvb = (const float*)d_in[4];
    const float* btab = (const float*)d_in[5];
    const float* pw   = (const float*)d_in[6];
    const float* pb   = (const float*)d_in[7];
    const float* n2w  = (const float*)d_in[8];
    const float* n2b  = (const float*)d_in[9];
    const float* w1   = (const float*)d_in[10];
    const float* b1   = (const float*)d_in[11];
    const float* w2   = (const float*)d_in[12];
    const float* b2   = (const float*)d_in[13];
    float* out = (float*)d_out;

    char* ws = (char*)d_ws;
    size_t off = 0;
    auto take = [&](size_t bytes) -> char* {
        char* p = ws + off;
        off = (off + bytes + 511) & ~(size_t)511;
        return p;
    };
    const size_t VT_BYTES = (size_t)NWIN * HEADS * HD * VPAD * sizeof(_Float16);
    _Float16* Xw = (_Float16*)take((size_t)MW * DIMc * 2);   // LN out / attn out / LN2 out
    _Float16* Qb = (_Float16*)take((size_t)MW * DIMc * 2);
    _Float16* Kb = (_Float16*)take((size_t)MW * DIMc * 2);
    _Float16* Vt = (_Float16*)take(VT_BYTES);
    float*    E  = (float*)take((size_t)MW * DIMc * 4);      // fp32 running activation
    _Float16* F  = (_Float16*)take((size_t)MW * 768 * 2);    // MLP hidden
    _Float16* Wq = (_Float16*)take((size_t)576 * DIMc * 2);
    _Float16* Wp = (_Float16*)take((size_t)DIMc * DIMc * 2);
    _Float16* W1 = (_Float16*)take((size_t)768 * DIMc * 2);
    _Float16* W2 = (_Float16*)take((size_t)DIMc * 768 * 2);

    for (int i = 0; i < 2; ++i) {
        int shifted = (i == 1);
        const float* qw_l = qkvw + (size_t)i * DIMc * 576;
        const float* pw_l = pw   + (size_t)i * DIMc * DIMc;
        const float* w1_l = w1   + (size_t)i * DIMc * 768;
        const float* w2_l = w2   + (size_t)i * 768 * DIMc;

        transpose_to_f16<<<dim3((576 * DIMc + 255) / 256), 256, 0, stream>>>(qw_l, Wq, DIMc, 576);
        transpose_to_f16<<<dim3((DIMc * DIMc + 255) / 256), 256, 0, stream>>>(pw_l, Wp, DIMc, DIMc);
        transpose_to_f16<<<dim3((DIMc * 768 + 255) / 256), 256, 0, stream>>>(w1_l, W1, DIMc, 768);
        transpose_to_f16<<<dim3((768 * DIMc + 255) / 256), 256, 0, stream>>>(w2_l, W2, 768, DIMc);
        hipMemsetAsync(Vt, 0, VT_BYTES, stream);

        const float* xin = (i == 0) ? x : E;
        // LN1 + roll + window partition -> f16
        ln_kernel<<<dim3(MW / 8), 256, 0, stream>>>(xin, n1w + i * DIMc, n1b + i * DIMc, Xw, 1, shifted);
        // QKV projection
        gemm_qkv<<<dim3(MW / 128, 576 / 64), 256, 0, stream>>>(Xw, Wq, qkvb + i * 576, Qb, Kb, Vt);
        // windowed attention (writes back into Xw)
        attn_kernel<<<dim3(9, HEADS, NWIN), 32, 0, stream>>>(
            Qb, Kb, Vt, Xw, btab + (size_t)i * 3312 * WT * HEADS, shifted);
        // output projection + un-roll + residual -> E (fp32)
        gemm_proj<<<dim3(MW / 128, DIMc / 64), 256, 0, stream>>>(Xw, Wp, pb + i * DIMc, xin, E, shifted);
        // LN2 -> f16 (token order)
        ln_kernel<<<dim3(MW / 8), 256, 0, stream>>>(E, n2w + i * DIMc, n2b + i * DIMc, Xw, 0, 0);
        // MLP
        gemm_gelu<<<dim3(MW / 128, 768 / 64), 256, 0, stream>>>(Xw, W1, b1 + i * 768, F);
        float* dst = (i == 1) ? out : E;
        gemm_resid<<<dim3(MW / 128, DIMc / 64), 256, 0, stream>>>(F, W2, b2 + i * DIMc, E, dst);
    }
}
